// GPTBlock_67834713473772
// MI455X (gfx1250) — compile-verified, matrix operand order checked by reading, not run on hardware
//
#include <hip/hip_runtime.h>
#include <hip/hip_bf16.h>

#define D_MODEL   1024
#define NUM_HEADS 16
#define HEAD_DIM  64
#define D_FF      4096
#define BATCH     4
#define SEQ       2048
#define ROWS      (BATCH * SEQ)   // 8192

typedef __attribute__((ext_vector_type(16))) __bf16 v16bf;
typedef __attribute__((ext_vector_type(8)))  float  v8f;
typedef __attribute__((ext_vector_type(4)))  int    v4i;

union Frag16 {
    v16bf v;
    unsigned int u[8];
    unsigned short h[16];
};

__device__ __forceinline__ unsigned short f2bf(float f) {
    unsigned int u = __float_as_uint(f);
    u += 0x7FFFu + ((u >> 16) & 1u);        // round-to-nearest-even
    return (unsigned short)(u >> 16);
}

// K index held by fragment element pair p for a lane in half `hi`
// (ISA 05_wmma §7.12.2, 16-bit A 16x32 layout; B symmetric per-column).
__device__ __forceinline__ int kpat(int p, int hi) {
    return 2 * p + ((p >= 4) ? 8 : 0) + 8 * hi;
}

__device__ __forceinline__ v8f v8f_zero() {
    v8f z;
#pragma unroll
    for (int i = 0; i < 8; i++) z[i] = 0.0f;
    return z;
}

// branch-free tanh-approx GELU: v * sigmoid(1.595769122*(v + 0.044715 v^3))
__device__ __forceinline__ float gelu_f(float v) {
    const float u = 1.5957691216f * (v + 0.044715f * v * v * v);
    return v / (1.0f + __expf(-u));
}

// ---------------------------------------------------------------------------
// CDNA5 async global->LDS path (double-buffer pipeline skeleton is shared;
// fallback degrades to synchronous copies).
// ---------------------------------------------------------------------------
#if defined(__gfx1250__) && \
    __has_builtin(__builtin_amdgcn_global_load_async_to_lds_b128) && \
    __has_builtin(__builtin_amdgcn_s_wait_asynccnt)
#define USE_ASYNC 1
#else
#define USE_ASYNC 0
#endif

#if USE_ASYNC
__device__ __forceinline__ void async_copy16(const void* g, void* l) {
    __builtin_amdgcn_global_load_async_to_lds_b128(
        (__attribute__((address_space(1))) v4i*)g,
        (__attribute__((address_space(3))) v4i*)l, 0, 0);
}
#define WAIT_ASYNC(n) __builtin_amdgcn_s_wait_asynccnt(n)
#else
__device__ __forceinline__ void async_copy16(const void* g, void* l) {
    *(uint4*)l = *(const uint4*)g;
}
#define WAIT_ASYNC(n) ((void)0)
#endif

// ---------------------------------------------------------------------------
// fp32 -> bf16 conversion (weights, once per launch)
// ---------------------------------------------------------------------------
__global__ __launch_bounds__(256)
void cvt_bf16_kernel(const float* __restrict__ in, unsigned short* __restrict__ out, int n) {
    int i = blockIdx.x * blockDim.x + threadIdx.x;
    if (i < n) out[i] = f2bf(in[i]);
}

// ---------------------------------------------------------------------------
// LayerNorm over last dim (1024), fp32 in -> bf16 out. One block per row.
// ---------------------------------------------------------------------------
__global__ __launch_bounds__(256)
void ln_bf16_kernel(const float* __restrict__ x,
                    const float* __restrict__ scale,
                    const float* __restrict__ bias,
                    unsigned short* __restrict__ out) {
    __shared__ float red[256];
    const int t = threadIdx.x;
    const size_t base = (size_t)blockIdx.x * D_MODEL;

    float4 v = *(const float4*)(x + base + t * 4);

    red[t] = v.x + v.y + v.z + v.w;
    __syncthreads();
#pragma unroll
    for (int off = 128; off > 0; off >>= 1) {
        if (t < off) red[t] += red[t + off];
        __syncthreads();
    }
    const float mu = red[0] * (1.0f / D_MODEL);
    __syncthreads();

    const float dx = v.x - mu, dy = v.y - mu, dz = v.z - mu, dw = v.w - mu;
    red[t] = dx * dx + dy * dy + dz * dz + dw * dw;
    __syncthreads();
#pragma unroll
    for (int off = 128; off > 0; off >>= 1) {
        if (t < off) red[t] += red[t + off];
        __syncthreads();
    }
    const float var = red[0] * (1.0f / D_MODEL);
    const float rs = rsqrtf(var + 1e-6f);

    const float xv[4] = {v.x, v.y, v.z, v.w};
#pragma unroll
    for (int i = 0; i < 4; i++) {
        const int c = t * 4 + i;
        out[base + c] = f2bf((xv[i] - mu) * rs * scale[c] + bias[c]);
    }
}

// ---------------------------------------------------------------------------
// bf16 WMMA GEMM: C[M,N] = A[M,K] @ B[K,N] + bias, compile-time epilogue EP:
//   EP 1: bf16 out      EP 2: gelu -> bf16 out      EP 3: +residual -> fp32 out
// Block tile 256x128, BK=32, 256 threads = 8 waves, each wave -> 64x64
// (16 WMMAs/k-step, B fragment reused across 4 A fragments).
// Double-buffered LDS fed by GLOBAL_LOAD_ASYNC_TO_LDS_B128 (6 ops/thread/tile).
// ---------------------------------------------------------------------------
#define BM 256
#define BN 128
#define BK 32
#define SA_STR 40    // ushorts: 80B rows (16B-aligned; 20-bank stride: conflict-free)
#define SB_STR 136   // ushorts: 272B rows (16B-aligned)

template <int EP>
__global__ __launch_bounds__(256)
void gemm_bf16_kernel(const unsigned short* __restrict__ A,
                      const unsigned short* __restrict__ B,
                      const float* __restrict__ bias,
                      const float* __restrict__ residual,
                      float* __restrict__ Cf,
                      unsigned short* __restrict__ Cb,
                      int M, int N, int K) {
    __shared__ unsigned short sA[2][BM][SA_STR];
    __shared__ unsigned short sB[2][BK][SB_STR];

    const int t = threadIdx.x;
    const int row0 = blockIdx.y * BM;
    const int col0 = blockIdx.x * BN;

    const int lane = t & 31;
    const int wid  = t >> 5;
    const int l16  = lane & 15;
    const int hi   = lane >> 4;
    const int wm = (wid & 3) * 64;   // wave row offset within block tile
    const int wn = (wid >> 2) * 64;  // wave col offset within block tile

    v8f acc[4][4];
#pragma unroll
    for (int i = 0; i < 4; i++)
#pragma unroll
        for (int j = 0; j < 4; j++) acc[i][j] = v8f_zero();

    // tile staging: A -> 4x b128 per thread (one 64B row), B -> 2x b128
    const int bk = t >> 3;            // B: k-row 0..31
    const int bn = (t & 7) * 16;      // B: n col 0..112 (32B span)

    const unsigned short* gA = A + (size_t)(row0 + t) * K;
    const unsigned short* gB = B + (size_t)bk * N + col0 + bn;

    const int nk = K / BK;
    // prime two tiles (6 async ops per thread per tile)
    {
#pragma unroll
        for (int c = 0; c < 4; c++)
            async_copy16(gA + c * 8, &sA[0][t][c * 8]);
        async_copy16(gB,     &sB[0][bk][bn]);
        async_copy16(gB + 8, &sB[0][bk][bn + 8]);
    }
    if (nk > 1) {
        const unsigned short* a1 = gA + BK;
        const unsigned short* b1 = gB + (size_t)BK * N;
#pragma unroll
        for (int c = 0; c < 4; c++)
            async_copy16(a1 + c * 8, &sA[1][t][c * 8]);
        async_copy16(b1,     &sB[1][bk][bn]);
        async_copy16(b1 + 8, &sB[1][bk][bn + 8]);
    }

    for (int i = 0; i < nk; ++i) {
        if (i + 1 < nk) WAIT_ASYNC(6);   // retire tile i (one tile in flight behind)
        else            WAIT_ASYNC(0);
        __syncthreads();

        const int buf = i & 1;
        const unsigned short (*sAb)[SA_STR] = sA[buf];
        const unsigned short (*sBb)[SB_STR] = sB[buf];

        Frag16 af[4];
#pragma unroll
        for (int mi = 0; mi < 4; mi++) {
            const int m = wm + mi * 16 + l16;
#pragma unroll
            for (int p = 0; p < 8; p++)
                af[mi].u[p] = *(const unsigned int*)&sAb[m][kpat(p, hi)];
        }
#pragma unroll
        for (int ni = 0; ni < 4; ni++) {
            const int n = wn + ni * 16 + l16;
            Frag16 bfr;
#pragma unroll
            for (int p = 0; p < 8; p++) {
                const int k = kpat(p, hi);
                const unsigned int lo = sBb[k][n];
                const unsigned int hi2 = sBb[k + 1][n];
                bfr.u[p] = lo | (hi2 << 16);
            }
#pragma unroll
            for (int mi = 0; mi < 4; mi++)
                acc[mi][ni] = __builtin_amdgcn_wmma_f32_16x16x32_bf16(
                    false, af[mi].v, false, bfr.v, (short)0, acc[mi][ni], false, false);
        }

        __syncthreads();  // all waves done reading buf before refilling it
        if (i + 2 < nk) {
            const unsigned short* an  = gA + (size_t)(i + 2) * BK;
            const unsigned short* bn2 = gB + (size_t)(i + 2) * BK * N;
#pragma unroll
            for (int c = 0; c < 4; c++)
                async_copy16(an + c * 8, &sA[buf][t][c * 8]);
            async_copy16(bn2,     &sB[buf][bk][bn]);
            async_copy16(bn2 + 8, &sB[buf][bk][bn + 8]);
        }
    }

    // ---- compile-time epilogue (branch-free) ----
#pragma unroll
    for (int mi = 0; mi < 4; mi++) {
#pragma unroll
        for (int ni = 0; ni < 4; ni++) {
            const int col = col0 + wn + ni * 16 + l16;
            const float bv = bias[col];
#pragma unroll
            for (int r = 0; r < 8; r++) {
                const int row = row0 + wm + mi * 16 + r + 8 * hi;
                const size_t idx = (size_t)row * N + col;
                const float v = acc[mi][ni][r] + bv;
                if constexpr (EP == 1) {
                    Cb[idx] = f2bf(v);
                } else if constexpr (EP == 2) {
                    Cb[idx] = f2bf(gelu_f(v));
                } else {
                    Cf[idx] = v + residual[idx];
                }
            }
        }
    }
}

// ---------------------------------------------------------------------------
// Causal flash attention, one wave per (b, h, 16-query tile).
// qkv layout: [B, S, 3, H, HEAD_DIM] bf16.  out layout: [B, S, H*HEAD_DIM] bf16.
// Mask compare/select only on the single diagonal tail tile.
// ---------------------------------------------------------------------------
__device__ __forceinline__ void attn_tile(
    const unsigned short* __restrict__ qkv, unsigned short (*sP)[34],
    int b, int h, int q0, int l16, int hi, int j0, bool masked,
    const Frag16* qa, v8f* o, float* mrow, float* lrow) {

    v8f sc[2];
#pragma unroll
    for (int tc = 0; tc < 2; tc++) {
        const int key = j0 + tc * 16 + l16;
        const unsigned short* krow =
            qkv + ((size_t)(b * SEQ + key) * 3 + 1) * D_MODEL + h * HEAD_DIM;
        v8f a = v8f_zero();
#pragma unroll
        for (int half = 0; half < 2; half++) {
            Frag16 kb;
#pragma unroll
            for (int p = 0; p < 8; p++)
                kb.u[p] = *(const unsigned int*)(krow + 32 * half + kpat(p, hi));
            a = __builtin_amdgcn_wmma_f32_16x16x32_bf16(
                false, qa[half].v, false, kb.v, (short)0, a, false, false);
        }
        sc[tc] = a;
    }
#pragma unroll
    for (int tc = 0; tc < 2; tc++) {
        const int key = j0 + tc * 16 + l16;
#pragma unroll
        for (int r = 0; r < 8; r++) {
            const float v = sc[tc][r] * 0.125f;
            if (masked) {
                const int qr = q0 + r + 8 * hi;
                sc[tc][r] = (key <= qr) ? v : -1.0e10f;
            } else {
                sc[tc][r] = v;
            }
        }
    }
#pragma unroll
    for (int r = 0; r < 8; r++) {
        float mx = fmaxf(sc[0][r], sc[1][r]);
#pragma unroll
        for (int off = 1; off < 16; off <<= 1)
            mx = fmaxf(mx, __shfl_xor(mx, off, 32));
        const float mnew = fmaxf(mrow[r], mx);
        const float alpha = __expf(mrow[r] - mnew);
#pragma unroll
        for (int nt = 0; nt < 4; nt++) o[nt][r] *= alpha;
        const float p0 = __expf(sc[0][r] - mnew);
        const float p1 = __expf(sc[1][r] - mnew);
        sc[0][r] = p0; sc[1][r] = p1;
        float sm = p0 + p1;
#pragma unroll
        for (int off = 1; off < 16; off <<= 1)
            sm += __shfl_xor(sm, off, 32);
        lrow[r] = lrow[r] * alpha + sm;
        mrow[r] = mnew;
    }
    __syncthreads();
#pragma unroll
    for (int tc = 0; tc < 2; tc++)
#pragma unroll
        for (int r = 0; r < 8; r++)
            sP[r + 8 * hi][tc * 16 + l16] = f2bf(sc[tc][r]);
    __syncthreads();

    Frag16 pa;
#pragma unroll
    for (int p = 0; p < 8; p++)
        pa.u[p] = *(const unsigned int*)&sP[l16][kpat(p, hi)];

#pragma unroll
    for (int nt = 0; nt < 4; nt++) {
        const int d = nt * 16 + l16;
        Frag16 vb;
#pragma unroll
        for (int p = 0; p < 8; p++) {
            const int k = kpat(p, hi);
            const unsigned int v0 =
                qkv[((size_t)(b * SEQ + j0 + k) * 3 + 2) * D_MODEL + h * HEAD_DIM + d];
            const unsigned int v1 =
                qkv[((size_t)(b * SEQ + j0 + k + 1) * 3 + 2) * D_MODEL + h * HEAD_DIM + d];
            vb.u[p] = v0 | (v1 << 16);
        }
        o[nt] = __builtin_amdgcn_wmma_f32_16x16x32_bf16(
            false, pa.v, false, vb.v, (short)0, o[nt], false, false);
    }
}

__global__ __launch_bounds__(32)
void attn_kernel(const unsigned short* __restrict__ qkv,
                 unsigned short* __restrict__ out) {
    __shared__ unsigned short sP[16][34];

    const int lane = threadIdx.x & 31;
    const int l16 = lane & 15;
    const int hi  = lane >> 4;

    int bid = blockIdx.x;
    const int nqt = SEQ / 16;
    const int qt = bid % nqt; bid /= nqt;
    const int h  = bid % NUM_HEADS;
    const int b  = bid / NUM_HEADS;
    const int q0 = qt * 16;

    Frag16 qa[2];
    {
        const unsigned short* qrow =
            qkv + ((size_t)(b * SEQ + q0 + l16) * 3 + 0) * D_MODEL + h * HEAD_DIM;
#pragma unroll
        for (int half = 0; half < 2; half++)
#pragma unroll
            for (int p = 0; p < 8; p++)
                qa[half].u[p] = *(const unsigned int*)(qrow + 32 * half + kpat(p, hi));
    }

    v8f o[4];
#pragma unroll
    for (int i = 0; i < 4; i++) o[i] = v8f_zero();
    float mrow[8], lrow[8];
#pragma unroll
    for (int r = 0; r < 8; r++) { mrow[r] = -3.0e38f; lrow[r] = 0.0f; }

    const int jm = q0 & ~31;   // all tiles below jm need no causal mask
    for (int j0 = 0; j0 < jm; j0 += 32)
        attn_tile(qkv, sP, b, h, q0, l16, hi, j0, false, qa, o, mrow, lrow);
    attn_tile(qkv, sP, b, h, q0, l16, hi, jm, true, qa, o, mrow, lrow);

#pragma unroll
    for (int nt = 0; nt < 4; nt++)
#pragma unroll
        for (int r = 0; r < 8; r++) {
            const int row = q0 + r + 8 * hi;
            const float val = o[nt][r] / lrow[r];
            out[(size_t)(b * SEQ + row) * D_MODEL + h * HEAD_DIM + nt * 16 + l16] =
                f2bf(val);
        }
}

// ---------------------------------------------------------------------------
// Host-side orchestration
// ---------------------------------------------------------------------------
extern "C" void kernel_launch(void* const* d_in, const int* in_sizes, int n_in,
                              void* d_out, int out_size, void* d_ws, size_t ws_size,
                              hipStream_t stream) {
    const float* x         = (const float*)d_in[0];
    const float* w_qkv     = (const float*)d_in[1];
    const float* b_qkv     = (const float*)d_in[2];
    const float* w_out     = (const float*)d_in[3];
    const float* b_out     = (const float*)d_in[4];
    const float* w_fc1     = (const float*)d_in[5];
    const float* b_fc1     = (const float*)d_in[6];
    const float* w_fc2     = (const float*)d_in[7];
    const float* b_fc2     = (const float*)d_in[8];
    const float* ln1_scale = (const float*)d_in[9];
    const float* ln1_bias  = (const float*)d_in[10];
    const float* ln2_scale = (const float*)d_in[11];
    const float* ln2_bias  = (const float*)d_in[12];
    float* out = (float*)d_out;

    char* base = (char*)d_ws;
    size_t off = 0;
    auto carve = [&](size_t bytes) {
        char* p = base + off;
        off += (bytes + 255) & ~(size_t)255;
        return p;
    };
    const size_t N_QKV = (size_t)D_MODEL * 3 * D_MODEL;
    const size_t N_OUT = (size_t)D_MODEL * D_MODEL;
    const size_t N_FC1 = (size_t)D_MODEL * D_FF;
    const size_t N_FC2 = (size_t)D_FF * D_MODEL;
    const size_t N_X   = (size_t)ROWS * D_MODEL;
    const size_t N_H   = (size_t)ROWS * D_FF;

    unsigned short* wqkv_bf = (unsigned short*)carve(N_QKV * 2);
    unsigned short* wout_bf = (unsigned short*)carve(N_OUT * 2);
    unsigned short* wfc1_bf = (unsigned short*)carve(N_FC1 * 2);
    unsigned short* wfc2_bf = (unsigned short*)carve(N_FC2 * 2);
    unsigned short* ln1_bf  = (unsigned short*)carve(N_X * 2);
    unsigned short* qkv_bf  = (unsigned short*)carve((size_t)ROWS * 3 * D_MODEL * 2);
    unsigned short* attn_bf = (unsigned short*)carve(N_X * 2);
    float*          x1_f    = (float*)carve(N_X * 4);
    unsigned short* ln2_bf  = (unsigned short*)carve(N_X * 2);
    unsigned short* h_bf    = (unsigned short*)carve(N_H * 2);
    (void)ws_size; (void)n_in; (void)in_sizes; (void)out_size;

    cvt_bf16_kernel<<<(N_QKV + 255) / 256, 256, 0, stream>>>(w_qkv, wqkv_bf, (int)N_QKV);
    cvt_bf16_kernel<<<(N_OUT + 255) / 256, 256, 0, stream>>>(w_out, wout_bf, (int)N_OUT);
    cvt_bf16_kernel<<<(N_FC1 + 255) / 256, 256, 0, stream>>>(w_fc1, wfc1_bf, (int)N_FC1);
    cvt_bf16_kernel<<<(N_FC2 + 255) / 256, 256, 0, stream>>>(w_fc2, wfc2_bf, (int)N_FC2);

    ln_bf16_kernel<<<ROWS, 256, 0, stream>>>(x, ln1_scale, ln1_bias, ln1_bf);

    gemm_bf16_kernel<1><<<dim3(3 * D_MODEL / BN, ROWS / BM), 256, 0, stream>>>(
        ln1_bf, wqkv_bf, b_qkv, nullptr, nullptr, qkv_bf, ROWS, 3 * D_MODEL, D_MODEL);

    attn_kernel<<<BATCH * NUM_HEADS * (SEQ / 16), 32, 0, stream>>>(qkv_bf, attn_bf);

    gemm_bf16_kernel<3><<<dim3(D_MODEL / BN, ROWS / BM), 256, 0, stream>>>(
        attn_bf, wout_bf, b_out, x, x1_f, nullptr, ROWS, D_MODEL, D_MODEL);

    ln_bf16_kernel<<<ROWS, 256, 0, stream>>>(x1_f, ln2_scale, ln2_bias, ln2_bf);

    gemm_bf16_kernel<2><<<dim3(D_FF / BN, ROWS / BM), 256, 0, stream>>>(
        ln2_bf, wfc1_bf, b_fc1, nullptr, nullptr, h_bf, ROWS, D_FF, D_MODEL);

    gemm_bf16_kernel<3><<<dim3(D_MODEL / BN, ROWS / BM), 256, 0, stream>>>(
        h_bf, wfc2_bf, b_fc2, x1_f, out, nullptr, ROWS, D_MODEL, D_FF);
}
